// MultiHeadAttention_36223754174786
// MI455X (gfx1250) — compile-verified, matrix operand order checked by reading, not run on hardware
//
#include <hip/hip_runtime.h>
#include <hip/hip_bf16.h>

// ---------------------------------------------------------------------------
// Transformer block for MI455X (gfx1250): bf16 WMMA GEMMs + flash attention.
// D = A(16x32 bf16) x B(32x16 bf16) + C(16x16 f32) via v_wmma_f32_16x16x32_bf16
// ---------------------------------------------------------------------------

typedef __attribute__((ext_vector_type(16))) __bf16 v16bf;
typedef __attribute__((ext_vector_type(8)))  float  v8f;

union Frag16 {
    v16bf v;
    uint4 q[2];
};

__device__ __forceinline__ unsigned short f2bf(float f) {
    union { float f; unsigned int u; } c;
    c.f = f;
    unsigned int u = c.u;
    u += 0x7fffu + ((u >> 16) & 1u);   // round-to-nearest-even
    return (unsigned short)(u >> 16);
}

__device__ __forceinline__ v8f wmma_bf16(const Frag16& a, const Frag16& b, v8f c) {
    return __builtin_amdgcn_wmma_f32_16x16x32_bf16(
        /*neg_a=*/false, a.v, /*neg_b=*/false, b.v,
        /*c_mod=*/(short)0, c, /*reuse_a=*/false, /*reuse_b=*/false);
}

// Problem constants
#define CB   2
#define CS   2048
#define CD   1024
#define CH   16
#define CDK  64
#define CFF  2048
#define CM   (CB * CS)   // 4096 rows of activations

// ---------------------------------------------------------------------------
// Tiled GEMM: out[M,N] = act( A[M,K] @ W[K,N] + bias ) (+ residual)
// Block tile 128x128, BK=32. 8 waves (4x2), each wave computes 32x64 of C.
// bf16 A path uses global_load_async_to_lds_b128 (ASYNCcnt) for staging.
// ---------------------------------------------------------------------------
template <bool A_BF16, bool RELU, bool RES, bool WF32, bool WB16>
__global__ __launch_bounds__(256)
void gemm_bias_kernel(const void* __restrict__ Ap,
                      const float* __restrict__ W,
                      const float* __restrict__ bias,
                      const float* __restrict__ res,
                      float* __restrict__ outf,
                      unsigned short* __restrict__ outb,
                      int M, int N, int K)
{
    __shared__ __align__(16) unsigned short As[128][32];    // [m][k]
    __shared__ __align__(16) unsigned short Bs[128][32];    // [n][k] (transposed)

    const int tid  = threadIdx.x;
    const int wave = tid >> 5;
    const int lane = tid & 31;
    const int half = lane >> 4;
    const int lm   = lane & 15;
    const int wm   = wave >> 1;           // 0..3  (M direction, 32 rows each)
    const int wn   = wave & 1;            // 0..1  (N direction, 64 cols each)
    const int m0   = blockIdx.y * 128;
    const int n0   = blockIdx.x * 128;

    v8f acc[2][4];
#pragma unroll
    for (int t = 0; t < 2; ++t)
#pragma unroll
        for (int u = 0; u < 4; ++u)
            acc[t][u] = (v8f){0.f, 0.f, 0.f, 0.f, 0.f, 0.f, 0.f, 0.f};

    for (int k0 = 0; k0 < K; k0 += 32) {
        __syncthreads();
        // ---- stage A tile ----
        if (A_BF16) {
            // A already bf16: async DMA straight into LDS (no VGPR round-trip)
            const unsigned short* A = (const unsigned short*)Ap;
#pragma unroll
            for (int i = 0; i < 2; ++i) {
                int e = (tid + i * 256) * 8;               // 8 bf16 (16B) per op
                int r = e >> 5, c = e & 31;
                unsigned long long ga =
                    (unsigned long long)&A[(size_t)(m0 + r) * K + k0 + c];
                unsigned lds = (unsigned)(unsigned long long)&As[r][c];
                asm volatile("global_load_async_to_lds_b128 %0, %1, off"
                             :: "v"(lds), "v"(ga) : "memory");
            }
        } else {
            const float* A = (const float*)Ap;
#pragma unroll
            for (int i = 0; i < 4; ++i) {
                int e = (tid + i * 256) * 4;               // 4 f32 per load
                int r = e >> 5, c = e & 31;
                float4 d = *(const float4*)&A[(size_t)(m0 + r) * K + k0 + c];
                As[r][c + 0] = f2bf(d.x);
                As[r][c + 1] = f2bf(d.y);
                As[r][c + 2] = f2bf(d.z);
                As[r][c + 3] = f2bf(d.w);
            }
        }
        // ---- stage W tile transposed: W[k][n] -> Bs[n][k] ----
#pragma unroll
        for (int i = 0; i < 4; ++i) {
            int e  = (tid + i * 256) * 4;
            int kk = e >> 7, nn = e & 127;
            float4 d = *(const float4*)&W[(size_t)(k0 + kk) * N + n0 + nn];
            Bs[nn + 0][kk] = f2bf(d.x);
            Bs[nn + 1][kk] = f2bf(d.y);
            Bs[nn + 2][kk] = f2bf(d.z);
            Bs[nn + 3][kk] = f2bf(d.w);
        }
        // prefetch next K tile into L2 while this tile computes
        if (k0 + 32 < K) {
            if (A_BF16) {
                const unsigned short* A = (const unsigned short*)Ap;
                int e = tid * 8, r = e >> 5, c = e & 31;
                __builtin_prefetch(&A[(size_t)(m0 + r) * K + k0 + 32 + c], 0, 3);
            } else {
                const float* A = (const float*)Ap;
                int e = tid * 4, r = e >> 5, c = e & 31;
                __builtin_prefetch(&A[(size_t)(m0 + r) * K + k0 + 32 + c], 0, 3);
            }
            int e = tid * 4, kk = e >> 7, nn = e & 127;
            __builtin_prefetch(&W[(size_t)(k0 + 32 + kk) * N + n0 + nn], 0, 3);
        }
        if (A_BF16)
            asm volatile("s_wait_asynccnt 0" ::: "memory");
        __syncthreads();

        // ---- fragments: A lanes hold rows, K split per half-wave ----
        Frag16 a[2], b[4];
#pragma unroll
        for (int t = 0; t < 2; ++t) {
            int row = wm * 32 + t * 16 + lm;
            a[t].q[0] = *(const uint4*)&As[row][half * 8];
            a[t].q[1] = *(const uint4*)&As[row][16 + half * 8];
        }
#pragma unroll
        for (int u = 0; u < 4; ++u) {
            int col = wn * 64 + u * 16 + lm;
            b[u].q[0] = *(const uint4*)&Bs[col][half * 8];
            b[u].q[1] = *(const uint4*)&Bs[col][16 + half * 8];
        }
#pragma unroll
        for (int t = 0; t < 2; ++t)
#pragma unroll
            for (int u = 0; u < 4; ++u)
                acc[t][u] = wmma_bf16(a[t], b[u], acc[t][u]);
    }

    // ---- epilogue: C layout -> lane holds (row = i + 8*half, col = lm) ----
#pragma unroll
    for (int u = 0; u < 4; ++u) {
        int col = n0 + wn * 64 + u * 16 + lm;
        float bv = bias[col];
#pragma unroll
        for (int t = 0; t < 2; ++t) {
#pragma unroll
            for (int i = 0; i < 8; ++i) {
                int row = m0 + wm * 32 + t * 16 + i + half * 8;
                float v = acc[t][u][i] + bv;
                if (RELU) v = fmaxf(v, 0.f);
                if (RES)  v += res[(size_t)row * N + col];
                if (WF32) outf[(size_t)row * N + col] = v;
                if (WB16) outb[(size_t)row * N + col] = f2bf(v);
            }
        }
    }
}

// ---------------------------------------------------------------------------
// Flash attention over bf16 Q/K/V (layout [B,S,D], head h owns cols h*64..+63)
// Block = 8 waves, each wave owns 16 query rows; 32-key tiles, online softmax.
// ---------------------------------------------------------------------------
__global__ __launch_bounds__(256)
void flash_attn_kernel(const unsigned short* __restrict__ qb,
                       const unsigned short* __restrict__ kb,
                       const unsigned short* __restrict__ vb,
                       float* __restrict__ attn)
{
    __shared__ __align__(16) unsigned short Vt[64][32];      // [dk][kk] V^T tile
    __shared__ __align__(16) unsigned short Ps[8][16][32];   // per-wave P tile

    const int tid  = threadIdx.x;
    const int wave = tid >> 5;
    const int lane = tid & 31;
    const int half = lane >> 4;
    const int lm   = lane & 15;
    const int b    = blockIdx.z;
    const int h    = blockIdx.y;
    const int q0   = blockIdx.x * 128 + wave * 16;
    const size_t rowbase = (size_t)b * CS;

    // Q fragments (K-dim = DK = 64 -> two bf16 WMMA K-steps), direct global b128
    Frag16 aq[2];
#pragma unroll
    for (int ks = 0; ks < 2; ++ks) {
        const unsigned short* p =
            qb + (rowbase + q0 + lm) * CD + h * CDK + ks * 32;
        aq[ks].q[0] = *(const uint4*)&p[half * 8];
        aq[ks].q[1] = *(const uint4*)&p[16 + half * 8];
    }

    float mrow[8], lrow[8];
    v8f o[4];
#pragma unroll
    for (int r = 0; r < 8; ++r) { mrow[r] = -1e30f; lrow[r] = 0.f; }
#pragma unroll
    for (int f = 0; f < 4; ++f)
        o[f] = (v8f){0.f, 0.f, 0.f, 0.f, 0.f, 0.f, 0.f, 0.f};

    const float scale = 0.125f;   // 1/sqrt(64)

    for (int kt = 0; kt < CS / 32; ++kt) {
        __syncthreads();   // previous iteration done with Vt
        // ---- cooperative stage of V^T tile: Vt[dk][kk], shared by 8 waves ----
        {
            int e  = tid * 8;
            int kk = e >> 6, dk = e & 63;
            uint4 d = *(const uint4*)&vb[(rowbase + kt * 32 + kk) * CD + h * CDK + dk];
            const unsigned short* ds = (const unsigned short*)&d;
#pragma unroll
            for (int j = 0; j < 8; ++j) Vt[dk + j][kk] = ds[j];
        }
        // prefetch next key/value tile
        if (kt + 1 < CS / 32) {
            int e  = tid * 8;
            int kk = e >> 6, dk = e & 63;
            __builtin_prefetch(&vb[(rowbase + (kt + 1) * 32 + kk) * CD + h * CDK + dk], 0, 3);
            __builtin_prefetch(&kb[(rowbase + (kt + 1) * 32 + kk) * CD + h * CDK + dk], 0, 3);
        }
        __syncthreads();

        // ---- S = Q @ K^T for 32 keys: prefetch 4 B-frags, then 4 WMMAs ----
        Frag16 bk[2][2];
#pragma unroll
        for (int u = 0; u < 2; ++u)
#pragma unroll
            for (int ks = 0; ks < 2; ++ks) {
                const unsigned short* p =
                    kb + (rowbase + kt * 32 + u * 16 + lm) * CD + h * CDK + ks * 32;
                bk[u][ks].q[0] = *(const uint4*)&p[half * 8];
                bk[u][ks].q[1] = *(const uint4*)&p[16 + half * 8];
            }
        v8f sf[2];
        sf[0] = (v8f){0.f, 0.f, 0.f, 0.f, 0.f, 0.f, 0.f, 0.f};
        sf[1] = sf[0];
#pragma unroll
        for (int u = 0; u < 2; ++u)
#pragma unroll
            for (int ks = 0; ks < 2; ++ks)
                sf[u] = wmma_bf16(aq[ks], bk[u][ks], sf[u]);

        // ---- online softmax; row r in VGPR r, lane halves hold rows r / r+8 ----
#pragma unroll
        for (int r = 0; r < 8; ++r) {
            float s0 = sf[0][r] * scale;
            float s1 = sf[1][r] * scale;
            float mx = fmaxf(s0, s1);
#pragma unroll
            for (int off = 1; off < 16; off <<= 1)
                mx = fmaxf(mx, __shfl_xor(mx, off, 32));
            float mnew = fmaxf(mrow[r], mx);
            float p0 = __expf(s0 - mnew);
            float p1 = __expf(s1 - mnew);
            float corr = __expf(mrow[r] - mnew);
            float sum = p0 + p1;
#pragma unroll
            for (int off = 1; off < 16; off <<= 1)
                sum += __shfl_xor(sum, off, 32);
            lrow[r] = lrow[r] * corr + sum;
            mrow[r] = mnew;
#pragma unroll
            for (int f = 0; f < 4; ++f) o[f][r] *= corr;
            // C-layout -> LDS P[row][col] for A-fragment reload
            Ps[wave][r + 8 * half][lm]      = f2bf(p0);
            Ps[wave][r + 8 * half][16 + lm] = f2bf(p1);
        }
        __syncthreads();   // P visible (also keeps waves in lockstep for Vt reuse)

        // ---- O += P @ V : one A frag (16x32), 4 B frags over DK ----
        Frag16 pa;
        pa.q[0] = *(const uint4*)&Ps[wave][lm][half * 8];
        pa.q[1] = *(const uint4*)&Ps[wave][lm][16 + half * 8];
        Frag16 bvf[4];
#pragma unroll
        for (int f = 0; f < 4; ++f) {
            bvf[f].q[0] = *(const uint4*)&Vt[f * 16 + lm][half * 8];
            bvf[f].q[1] = *(const uint4*)&Vt[f * 16 + lm][16 + half * 8];
        }
#pragma unroll
        for (int f = 0; f < 4; ++f)
            o[f] = wmma_bf16(pa, bvf[f], o[f]);
    }

    // ---- normalize and store fp32 attention output [B,S,D] ----
    float invl[8];
#pragma unroll
    for (int r = 0; r < 8; ++r) invl[r] = 1.f / lrow[r];
#pragma unroll
    for (int f = 0; f < 4; ++f) {
#pragma unroll
        for (int i = 0; i < 8; ++i) {
            int row = q0 + i + 8 * half;
            attn[(rowbase + row) * CD + h * CDK + f * 16 + lm] = o[f][i] * invl[i];
        }
    }
}

// ---------------------------------------------------------------------------
// LayerNorm over D=1024; writes fp32 (residual source) + bf16 (FF1 input)
// ---------------------------------------------------------------------------
__global__ __launch_bounds__(256)
void layernorm_kernel(const float* __restrict__ x,
                      const float* __restrict__ g,
                      const float* __restrict__ bta,
                      float* __restrict__ ffi,
                      unsigned short* __restrict__ ffib)
{
    __shared__ float red[2][8];
    const int row = blockIdx.x;
    const float* xr = x + (size_t)row * CD;

    float vals[4];
    float s = 0.f, ss = 0.f;
#pragma unroll
    for (int i = 0; i < 4; ++i) {
        vals[i] = xr[threadIdx.x + i * 256];
        s  += vals[i];
        ss += vals[i] * vals[i];
    }
#pragma unroll
    for (int off = 1; off < 32; off <<= 1) {
        s  += __shfl_xor(s,  off, 32);
        ss += __shfl_xor(ss, off, 32);
    }
    const int wave = threadIdx.x >> 5, lane = threadIdx.x & 31;
    if (lane == 0) { red[0][wave] = s; red[1][wave] = ss; }
    __syncthreads();
    float ts = 0.f, tss = 0.f;
#pragma unroll
    for (int w = 0; w < 8; ++w) { ts += red[0][w]; tss += red[1][w]; }
    const float mu  = ts * (1.f / CD);
    const float var = tss * (1.f / CD) - mu * mu;
    const float inv = rsqrtf(var + 1e-5f);
#pragma unroll
    for (int i = 0; i < 4; ++i) {
        int c = threadIdx.x + i * 256;
        float y = (vals[i] - mu) * inv * g[c] + bta[c];
        ffi[(size_t)row * CD + c]  = y;
        ffib[(size_t)row * CD + c] = f2bf(y);
    }
}

// ---------------------------------------------------------------------------
// Launch pipeline
// ---------------------------------------------------------------------------
extern "C" void kernel_launch(void* const* d_in, const int* in_sizes, int n_in,
                              void* d_out, int out_size, void* d_ws, size_t ws_size,
                              hipStream_t stream)
{
    (void)in_sizes; (void)n_in; (void)out_size; (void)ws_size;

    const float* query = (const float*)d_in[0];
    const float* key   = (const float*)d_in[1];
    const float* value = (const float*)d_in[2];
    const float* Wq    = (const float*)d_in[3];
    const float* bq    = (const float*)d_in[4];
    const float* Wk    = (const float*)d_in[5];
    const float* bk    = (const float*)d_in[6];
    const float* Wv    = (const float*)d_in[7];
    const float* bv    = (const float*)d_in[8];
    const float* ln_g  = (const float*)d_in[9];
    const float* ln_b  = (const float*)d_in[10];
    const float* W1    = (const float*)d_in[11];
    const float* b1    = (const float*)d_in[12];
    const float* W2    = (const float*)d_in[13];
    const float* b2    = (const float*)d_in[14];

    // workspace carve-up (256B aligned chunks)
    size_t off = 0;
    auto carve = [&](size_t bytes) -> void* {
        void* p = (char*)d_ws + off;
        off += (bytes + 255) & ~(size_t)255;
        return p;
    };
    unsigned short* qb   = (unsigned short*)carve((size_t)CM * CD * 2);
    unsigned short* kbuf = (unsigned short*)carve((size_t)CM * CD * 2);
    unsigned short* vbuf = (unsigned short*)carve((size_t)CM * CD * 2);
    float*          attn = (float*)carve((size_t)CM * CD * 4);
    float*          ffi  = (float*)carve((size_t)CM * CD * 4);
    unsigned short* ffib = (unsigned short*)carve((size_t)CM * CD * 2);
    unsigned short* hb   = (unsigned short*)carve((size_t)CM * CFF * 2);

    const dim3 blk(256);

    // QKV projections -> bf16 activations
    gemm_bias_kernel<false, false, false, false, true>
        <<<dim3(CD / 128, CM / 128), blk, 0, stream>>>(
            query, Wq, bq, nullptr, nullptr, qb, CM, CD, CD);
    gemm_bias_kernel<false, false, false, false, true>
        <<<dim3(CD / 128, CM / 128), blk, 0, stream>>>(
            key, Wk, bk, nullptr, nullptr, kbuf, CM, CD, CD);
    gemm_bias_kernel<false, false, false, false, true>
        <<<dim3(CD / 128, CM / 128), blk, 0, stream>>>(
            value, Wv, bv, nullptr, nullptr, vbuf, CM, CD, CD);

    // flash attention
    flash_attn_kernel<<<dim3(CS / 128, CH, CB), blk, 0, stream>>>(
        qb, kbuf, vbuf, attn);

    // LayerNorm
    layernorm_kernel<<<dim3(CM), blk, 0, stream>>>(attn, ln_g, ln_b, ffi, ffib);

    // FF1: h = relu(ffi @ W1 + b1), bf16 out (A staged via async-to-LDS)
    gemm_bias_kernel<true, true, false, false, true>
        <<<dim3(CFF / 128, CM / 128), blk, 0, stream>>>(
            ffib, W1, b1, nullptr, nullptr, hb, CM, CFF, CD);

    // FF2: out = ffi + (h @ W2 + b2), fp32 out (A staged via async-to-LDS)
    gemm_bias_kernel<true, false, true, true, false>
        <<<dim3(CD / 128, CM / 128), blk, 0, stream>>>(
            hb, W2, b2, ffi, (float*)d_out, nullptr, CM, CD, CFF);
}
